// WaveletGatedNet_74715251081588
// MI455X (gfx1250) — matrix operand order repaired and checked by reading, not compiled, not run on hardware
//
#include <hip/hip_runtime.h>

// ---------------------------------------------------------------------------
// WaveletGatedNet for MI455X (gfx1250, wave32, WMMA + async LDS loads)
//   B=128, C=16, L=4096, LEVEL=3 -> D=512, bands=8, M = B*C*8 = 16384
//   pipeline: WPT -> (coeffs @ W^T) WMMA f16/f32 -> sigmoid gate -> threshold
//             -> IWPT -> + x
// ---------------------------------------------------------------------------

typedef __attribute__((ext_vector_type(16))) _Float16 v16h;
typedef __attribute__((ext_vector_type(8)))  _Float16 v8h;
typedef __attribute__((ext_vector_type(8)))  float    v8f;

#define BC_PAIRS 2048         // B*C
#define SEQ_LEN  4096         // L
#define DD       512          // D = L / 8
#define NBAND    8
#define M_TOT    16384        // BC_PAIRS * NBAND
#define S3       0.35355339059327373f   // (1/sqrt(2))^3

// ---------------------------------------------------------------------------
// Kernel 0: W (f32, row-major [e][d]) -> f16 copy for WMMA B-matrix
// ---------------------------------------------------------------------------
__global__ __launch_bounds__(256)
void wcvt_kernel(const float* __restrict__ W, _Float16* __restrict__ Wh)
{
    const int i = blockIdx.x * 256 + threadIdx.x;   // 262144 elements exactly
    Wh[i] = (_Float16)W[i];
}

// ---------------------------------------------------------------------------
// Kernel 1: 3-level Haar WPT. One thread per 8-sample group.
// ---------------------------------------------------------------------------
__global__ __launch_bounds__(256)
void wpt_kernel(const float* __restrict__ x,
                float* __restrict__ cf32, _Float16* __restrict__ cf16)
{
    const int t  = blockIdx.x * 256 + threadIdx.x;   // 0 .. 2048*512-1
    const int bc = t >> 9;
    const int g  = t & 511;

    const float4* xp = (const float4*)(x + ((size_t)bc << 12) + g * 8);
    const float4 u = xp[0];
    const float4 v = xp[1];

    // level 1
    const float a0 = u.x + u.y, d0 = u.x - u.y;
    const float a1 = u.z + u.w, d1 = u.z - u.w;
    const float a2 = v.x + v.y, d2 = v.x - v.y;
    const float a3 = v.z + v.w, d3 = v.z - v.w;
    // level 2
    const float aa0 = a0 + a1, ad0 = a0 - a1, aa1 = a2 + a3, ad1 = a2 - a3;
    const float da0 = d0 + d1, dd0 = d0 - d1, da1 = d2 + d3, dd1 = d2 - d3;
    // level 3 (band order: aaa aad ada add daa dad dda ddd)
    float o[NBAND];
    o[0] = (aa0 + aa1) * S3;  o[1] = (aa0 - aa1) * S3;
    o[2] = (ad0 + ad1) * S3;  o[3] = (ad0 - ad1) * S3;
    o[4] = (da0 + da1) * S3;  o[5] = (da0 - da1) * S3;
    o[6] = (dd0 + dd1) * S3;  o[7] = (dd0 - dd1) * S3;

    const size_t base = ((size_t)bc * NBAND) * DD + g;
#pragma unroll
    for (int n = 0; n < NBAND; ++n) {
        cf32[base + n * DD] = o[n];
        cf16[base + n * DD] = (_Float16)o[n];
    }
}

// ---------------------------------------------------------------------------
// Async memory->LDS 16-byte copy (per lane), tracked by ASYNCcnt.
// VDST operand = 32-bit LDS byte address (low 32 bits of the generic
// __shared__ pointer), VADDR = 64-bit global address (GV mode).
// ---------------------------------------------------------------------------
__device__ __forceinline__ void async_ld_b128(void* lds_ptr, const void* gptr)
{
    const unsigned lds_addr = (unsigned)(unsigned long long)lds_ptr;
    asm volatile("global_load_async_to_lds_b128 %0, %1, off"
                 :
                 : "v"(lds_addr), "v"((unsigned long long)gptr)
                 : "memory");
}

__device__ __forceinline__ void wait_asynccnt0()
{
    asm volatile("s_wait_asynccnt 0x0" ::: "memory");
}

// ---------------------------------------------------------------------------
// Kernel 2: logits = A(16384x512, f16) * W^T with v_wmma_f32_16x16x32_f16,
// epilogue: sigmoid gate * coeff, threshold, in place over f32 coeffs.
//
// Block tile 128(M) x 128(N), 256 threads = 8 wave32; wave tile 32x64
// (2 M-frags x 4 N-frags). K streamed in 32-wide steps through
// double-buffered LDS, fed by global_load_async_to_lds_b128 (ASYNCcnt),
// one workgroup barrier per K-step.
// ---------------------------------------------------------------------------
__global__ __launch_bounds__(256)
void gemm_gate_kernel(const _Float16* __restrict__ A,    // cf16 [M_TOT][DD]
                      const _Float16* __restrict__ Bw,   // W f16 [DD][DD] row=e
                      const float* __restrict__ bias,    // [DD]
                      const float* __restrict__ thrp,    // scalar
                      float* __restrict__ CG)            // f32 coeffs -> gated
{
    __shared__ _Float16 sA[2][128][40];   // [buf][m][k], stride 40 halves (80B)
    __shared__ _Float16 sB[2][128][40];   // [buf][e][k]

    const int t    = threadIdx.x;
    const int wave = t >> 5;
    const int lane = t & 31;
    const int wm   = wave & 3;          // 0..3 -> M offset wm*32
    const int wn   = wave >> 2;         // 0..1 -> N offset wn*64
    const int lh   = lane >> 4;         // half-wave select
    const int ll   = lane & 15;

    const int mBase = blockIdx.x * 128;
    const int nBase = blockIdx.y * 128;

    // staging: each thread moves two 16B chunks of A and two of B per K-step
    const int c0 = t, c1 = t + 256;
    const int ar0 = c0 >> 2, ak0 = (c0 & 3) * 8;
    const int ar1 = c1 >> 2, ak1 = (c1 & 3) * 8;

    const _Float16* gA0 = A  + (size_t)(mBase + ar0) * DD + ak0;
    const _Float16* gA1 = A  + (size_t)(mBase + ar1) * DD + ak1;
    const _Float16* gB0 = Bw + (size_t)(nBase + ar0) * DD + ak0;
    const _Float16* gB1 = Bw + (size_t)(nBase + ar1) * DD + ak1;

    v8f acc[2][4];
#pragma unroll
    for (int i = 0; i < 2; ++i)
#pragma unroll
        for (int j = 0; j < 4; ++j) acc[i][j] = (v8f)(0.0f);

    // issue async loads of one K-tile into LDS buffer `buf`
    auto issue_tile = [&](int kt, int buf) {
        const int kb = kt * 32;
        async_ld_b128(&sA[buf][ar0][ak0], gA0 + kb);
        async_ld_b128(&sA[buf][ar1][ak1], gA1 + kb);
        async_ld_b128(&sB[buf][ar0][ak0], gB0 + kb);
        async_ld_b128(&sB[buf][ar1][ak1], gB1 + kb);
    };

    issue_tile(0, 0);

    for (int kt = 0; kt < DD / 32; ++kt) {
        const int cur = kt & 1;

        wait_asynccnt0();      // this wave's loads for buf[cur] have landed
        __syncthreads();       // ... and every other wave's too

        if (kt < DD / 32 - 1)  // overlap next tile's loads with this compute;
            issue_tile(kt + 1, cur ^ 1);   // buf[cur^1] was last read before
                                           // the barrier above -> no race
        // A fragments: 16x32 f16. lanes 0-15: row=ll, K chunks {0-7,16-23};
        // lanes 16-31: row=ll, K chunks {8-15,24-31}.
        v16h af[2];
#pragma unroll
        for (int fm = 0; fm < 2; ++fm) {
            const int mr = wm * 32 + fm * 16 + ll;
            v8h lo = *(const v8h*)&sA[cur][mr][lh * 8];
            v8h hi = *(const v8h*)&sA[cur][mr][16 + lh * 8];
            af[fm] = __builtin_shufflevector(lo, hi,
                       0,1,2,3,4,5,6,7,8,9,10,11,12,13,14,15);
        }
        // B fragments: 32x16 f16. lane holds one column (e fixed),
        // lanes 0-15: K chunks {0-7,16-23}; lanes 16-31: K chunks {8-15,24-31}.
        v16h bf[4];
#pragma unroll
        for (int fn = 0; fn < 4; ++fn) {
            const int er = wn * 64 + fn * 16 + ll;
            v8h lo = *(const v8h*)&sB[cur][er][lh * 8];
            v8h hi = *(const v8h*)&sB[cur][er][16 + lh * 8];
            bf[fn] = __builtin_shufflevector(lo, hi,
                       0,1,2,3,4,5,6,7,8,9,10,11,12,13,14,15);
        }

#pragma unroll
        for (int fm = 0; fm < 2; ++fm)
#pragma unroll
            for (int fn = 0; fn < 4; ++fn)
                acc[fm][fn] = __builtin_amdgcn_wmma_f32_16x16x32_f16(
                    false, af[fm], false, bf[fn],
                    (short)0, acc[fm][fn], false, false);
    }

    // Epilogue: D layout -> VGPR j, lanes 0-15: M=j, N=ll; lanes 16-31: M=j+8.
    const float thr = *thrp;
#pragma unroll
    for (int fm = 0; fm < 2; ++fm) {
#pragma unroll
        for (int fn = 0; fn < 4; ++fn) {
            const int m0 = mBase + wm * 32 + fm * 16 + lh * 8;
            const int e  = nBase + wn * 64 + fn * 16 + ll;
            const float be = bias[e];
#pragma unroll
            for (int j = 0; j < 8; ++j) {
                const float logit = acc[fm][fn][j] + be;
                const float gate  = 1.0f / (1.0f + __expf(-logit));
                const size_t idx  = (size_t)(m0 + j) * DD + e;
                float gv = CG[idx] * gate;
                gv = (__builtin_fabsf(gv) < thr) ? 0.0f : gv;
                CG[idx] = gv;
            }
        }
    }
}

// ---------------------------------------------------------------------------
// Kernel 3: inverse WPT of gated coeffs + residual x
// ---------------------------------------------------------------------------
__global__ __launch_bounds__(256)
void iwpt_kernel(const float* __restrict__ gated,
                 const float* __restrict__ x, float* __restrict__ out)
{
    const int t  = blockIdx.x * 256 + threadIdx.x;
    const int bc = t >> 9;
    const int g  = t & 511;

    const size_t base = ((size_t)bc * NBAND) * DD + g;
    const float o0 = gated[base + 0 * DD], o1 = gated[base + 1 * DD];
    const float o2 = gated[base + 2 * DD], o3 = gated[base + 3 * DD];
    const float o4 = gated[base + 4 * DD], o5 = gated[base + 5 * DD];
    const float o6 = gated[base + 6 * DD], o7 = gated[base + 7 * DD];

    // undo level 3
    const float aa0 = o0 + o1, aa1 = o0 - o1, ad0 = o2 + o3, ad1 = o2 - o3;
    const float da0 = o4 + o5, da1 = o4 - o5, dd0 = o6 + o7, dd1 = o6 - o7;
    // undo level 2
    const float a0 = aa0 + ad0, a1 = aa0 - ad0, a2 = aa1 + ad1, a3 = aa1 - ad1;
    const float d0 = da0 + dd0, d1 = da0 - dd0, d2 = da1 + dd1, d3 = da1 - dd1;

    const size_t xoff = ((size_t)bc << 12) + g * 8;
    const float4* xp = (const float4*)(x + xoff);
    const float4 u = xp[0];
    const float4 v = xp[1];

    float4 r0, r1;
    r0.x = (a0 + d0) * S3 + u.x;  r0.y = (a0 - d0) * S3 + u.y;
    r0.z = (a1 + d1) * S3 + u.z;  r0.w = (a1 - d1) * S3 + u.w;
    r1.x = (a2 + d2) * S3 + v.x;  r1.y = (a2 - d2) * S3 + v.y;
    r1.z = (a3 + d3) * S3 + v.z;  r1.w = (a3 - d3) * S3 + v.w;

    float4* op = (float4*)(out + xoff);
    op[0] = r0;
    op[1] = r1;
}

// ---------------------------------------------------------------------------
extern "C" void kernel_launch(void* const* d_in, const int* in_sizes, int n_in,
                              void* d_out, int out_size, void* d_ws, size_t ws_size,
                              hipStream_t stream)
{
    (void)in_sizes; (void)n_in; (void)out_size; (void)ws_size;

    const float* x    = (const float*)d_in[0];   // [128,16,4096]
    const float* W    = (const float*)d_in[1];   // [512,512]
    const float* bias = (const float*)d_in[2];   // [512]
    const float* thr  = (const float*)d_in[3];   // scalar
    float* out = (float*)d_out;

    char* ws = (char*)d_ws;
    float*    cf32 = (float*)ws;                                    // 32 MiB (coeffs -> gated, in place)
    _Float16* cf16 = (_Float16*)(ws + (size_t)32 * 1024 * 1024);    // 16 MiB
    _Float16* wh   = (_Float16*)(ws + (size_t)48 * 1024 * 1024);    // 0.5 MiB

    wcvt_kernel<<<dim3(1024), dim3(256), 0, stream>>>(W, wh);
    wpt_kernel<<<dim3(4096), dim3(256), 0, stream>>>(x, cf32, cf16);
    gemm_gate_kernel<<<dim3(M_TOT / 128, DD / 128), dim3(256), 0, stream>>>(
        cf16, wh, bias, thr, cf32);
    iwpt_kernel<<<dim3(4096), dim3(256), 0, stream>>>(cf32, x, out);
}
